// RecursiveNet_86990267613488
// MI455X (gfx1250) — compile-verified, matrix operand order checked by reading, not run on hardware
//
#include <hip/hip_runtime.h>
#include <hip/hip_bf16.h>

// ---------------------------------------------------------------------------
// RecursiveNet on MI455X (gfx1250, wave32, WMMA).
// Memory-bound: bucket edges/nodes by depth so each is touched exactly once.
// Dense micro-GEMMs on v_wmma_f32_16x16x32_f16 (16-row tile per wave).
// Weight tiles are pre-packed into native wave32 B-fragment layout once, so
// hot kernels load each fragment as 32B/lane vector loads (L2 resident).
// leaky_relu(x) == max(x, 0.01x): v_mul + v_max_num, no cmp/cndmask chain.
// ---------------------------------------------------------------------------

typedef __attribute__((ext_vector_type(16))) _Float16 v16h;
typedef __attribute__((ext_vector_type(8)))  _Float16 v8h;
typedef __attribute__((ext_vector_type(8)))  float    v8f;

#define ATOM_F 14
#define BOND_F 3
#define INNER_F 20
#define HID_F 34
#define REC_F 20
#define OHID_F 25
#define MAXD 16

// Fragment table indices (19 fragments total, packed by pack_frags):
#define F_INNER  0   // inner_w  [23x20]: nt 0..1
#define F_N0W1   2   // net0_w1  [14x34]: nt 0..2
#define F_N0W2   5   // net0_w2  [34x20]: kt*2+nt (kt 0..1, nt 0..1)
#define F_NW1    9   // net_w1   [34x34]: kt*3+nt (kt 0..1, nt 0..2)
#define F_NW2   15   // net_w2   [34x20]: kt*2+nt
#define N_FRAGS 19
#define STG_W   72   // stage row width (144B: 16B-aligned chunks, conflict-free)

// Exact leaky-relu with slope 0.01: max(x, 0.01x) (identical for all x).
__device__ __forceinline__ float lrelu(float x) { return fmaxf(x, 0.01f * x); }
__device__ __forceinline__ int   imin(int a, int b) { return a < b ? a : b; }

__device__ __forceinline__ v8f wmma32(v16h a, v16h b, v8f c) {
  return __builtin_amdgcn_wmma_f32_16x16x32_f16(false, a, false, b, (short)0, c,
                                                false, false);
}

// Build a B fragment from row-major fp32 W[K x N], tile at (k0, n0).
// B layout: lane<16 -> N=lane, K={0..7,16..23}; lane>=16 -> N=lane-16,
// K={8..15,24..31}. Out-of-range K/N -> 0.
__device__ __forceinline__ v16h bfrag(const float* __restrict__ W, int K, int N,
                                      int k0, int n0, int lane) {
  const int n = n0 + (lane & 15);
  const int khalf = (lane >> 4) << 3;
  v16h r;
#pragma unroll
  for (int j = 0; j < 16; ++j) {
    const int k = k0 + khalf + j + ((j & 8) ? 8 : 0);
    float v = 0.f;
    if (k < K && n < N) v = W[k * N + n];
    r[j] = (_Float16)v;
  }
  return r;
}

__device__ __forceinline__ v16h ldfrag(const _Float16* __restrict__ frags, int f,
                                       int lane) {
  return *(const v16h*)(frags + ((size_t)f * 32 + lane) * 16);
}

__device__ __forceinline__ v16h cat8(v8h lo, v8h hi) {
  return __builtin_shufflevector(lo, hi, 0, 1, 2, 3, 4, 5, 6, 7,
                                 8, 9, 10, 11, 12, 13, 14, 15);
}

// One-time: bake every weight tile into its wave32 B-fragment image.
__global__ void pack_frags(const float* __restrict__ iw, const float* __restrict__ w01,
                           const float* __restrict__ w02, const float* __restrict__ w1,
                           const float* __restrict__ w2, _Float16* __restrict__ frags) {
  const int lane = threadIdx.x & 31;
  const int f = blockIdx.x;
  v16h r;
  if (f < F_N0W1) {                    // inner_w
    r = bfrag(iw, BOND_F + REC_F, INNER_F, 0, (f - F_INNER) * 16, lane);
  } else if (f < F_N0W2) {             // net0_w1
    r = bfrag(w01, ATOM_F, HID_F, 0, (f - F_N0W1) * 16, lane);
  } else if (f < F_NW1) {              // net0_w2
    const int i = f - F_N0W2;
    r = bfrag(w02, HID_F, REC_F, (i >> 1) * 32, (i & 1) * 16, lane);
  } else if (f < F_NW2) {              // net_w1
    const int i = f - F_NW1;
    r = bfrag(w1, HID_F, HID_F, (i / 3) * 32, (i % 3) * 16, lane);
  } else {                             // net_w2
    const int i = f - F_NW2;
    r = bfrag(w2, HID_F, REC_F, (i >> 1) * 32, (i & 1) * 16, lane);
  }
  *(v16h*)(frags + ((size_t)f * 32 + lane) * 16) = r;
}

// ---------------------------------------------------------------------------
// Utility / bucketing kernels
// ---------------------------------------------------------------------------
__global__ void zero_f32(float* p, long long n) {
  long long i = (long long)blockIdx.x * blockDim.x + threadIdx.x;
  const long long s = (long long)gridDim.x * blockDim.x;
  for (; i < n; i += s) p[i] = 0.f;
}
__global__ void zero_i32(int* p, long long n) {
  long long i = (long long)blockIdx.x * blockDim.x + threadIdx.x;
  const long long s = (long long)gridDim.x * blockDim.x;
  for (; i < n; i += s) p[i] = 0;
}
__global__ void mark_has_child(const int* __restrict__ parent, int* __restrict__ hc, int E) {
  const int e = blockIdx.x * blockDim.x + threadIdx.x;
  if (e < E) hc[parent[e]] = 1;  // benign race, all writers store 1
}
__global__ void count_edges(const int* __restrict__ parent, const int* __restrict__ depth,
                            int* __restrict__ ectr, int E) {
  const int e = blockIdx.x * blockDim.x + threadIdx.x;
  if (e < E) atomicAdd(&ectr[depth[parent[e]]], 1);
}
__global__ void count_nodes(const int* __restrict__ depth, const int* __restrict__ hc,
                            int* __restrict__ nctr, int N) {
  const int n = blockIdx.x * blockDim.x + threadIdx.x;
  if (n < N && hc[n]) atomicAdd(&nctr[depth[n]], 1);
}
// ctr layout: [0..15]=cnt, [16..31]=off, [32..47]=cursor
__global__ void prefix16(int* ctr) {
  if (threadIdx.x == 0 && blockIdx.x == 0) {
    int s = 0;
    for (int d = 0; d < MAXD; ++d) { ctr[16 + d] = s; ctr[32 + d] = s; s += ctr[d]; }
  }
}
__global__ void scatter_edges(const int* __restrict__ parent, const int* __restrict__ depth,
                              int* __restrict__ ectr, int* __restrict__ ebkt, int E) {
  const int e = blockIdx.x * blockDim.x + threadIdx.x;
  if (e < E) {
    const int pos = atomicAdd(&ectr[32 + depth[parent[e]]], 1);
    ebkt[pos] = e;
  }
}
__global__ void scatter_nodes(const int* __restrict__ depth, const int* __restrict__ hc,
                              int* __restrict__ nctr, int* __restrict__ nbkt, int N) {
  const int n = blockIdx.x * blockDim.x + threadIdx.x;
  if (n < N && hc[n]) {
    const int pos = atomicAdd(&nctr[32 + depth[n]], 1);
    nbkt[pos] = n;
  }
}

// ---------------------------------------------------------------------------
// Stage helpers: pad cols [48,64) with zeros (cols >= HID_F are already 0 by
// construction), so layer-2 A fragments are four mask-free ds_load_b128.
// ---------------------------------------------------------------------------
__device__ __forceinline__ void stage_pad_zero(_Float16 (*stage)[STG_W], int lane) {
  v8h z = {};
  *(v8h*)(&stage[lane & 15][48 + (((lane >> 4)) << 3)]) = z;
}
__device__ __forceinline__ void stage_read(const _Float16 (*stage)[STG_W], int m,
                                           int khalf, v16h& y0, v16h& y1) {
  const _Float16* srow = &stage[m][0];
  const v8h c0 = *(const v8h*)(srow + khalf);
  const v8h c1 = *(const v8h*)(srow + 16 + khalf);
  const v8h c2 = *(const v8h*)(srow + 32 + khalf);
  const v8h c3 = *(const v8h*)(srow + 48 + khalf);
  y0 = cat8(c0, c1);
  y1 = cat8(c2, c3);
}

// ---------------------------------------------------------------------------
// Leaf MLP (all nodes): h = leaky(leaky(atom @ w1 + b1) @ w2 + b2)
// 4 waves/block, 16 nodes per wave tile.
// ---------------------------------------------------------------------------
__global__ void __launch_bounds__(128) leaf_kernel(
    const float* __restrict__ atom, const _Float16* __restrict__ frags,
    const float* __restrict__ b1, const float* __restrict__ b2,
    float* __restrict__ h, int n_nodes) {
  __shared__ _Float16 stage[4][16][STG_W];
  const int tiles = (n_nodes + 15) >> 4;
  if ((int)blockIdx.x * 4 >= tiles) return;  // uniform block exit
  const int lane = threadIdx.x & 31;
  const int wave = threadIdx.x >> 5;
  const int tile = blockIdx.x * 4 + wave;
  const bool wv = tile < tiles;
  const int base = (wv ? tile : 0) << 4;
  const int m = lane & 15;
  const int khalf = (lane >> 4) << 3;
  const int row = imin(base + m, n_nodes - 1);

  // A fragment: K = ATOM_F (14), one K tile.
  v16h a;
#pragma unroll
  for (int j = 0; j < 16; ++j) {
    const int k = khalf + j + ((j & 8) ? 8 : 0);
    a[j] = (_Float16)((k < ATOM_F) ? atom[row * ATOM_F + k] : 0.f);
  }
  stage_pad_zero(stage[wave], lane);
  // Layer 1: N = 34 (3 tiles) -> LDS stage (cols >= 34 are exactly 0).
#pragma unroll
  for (int nt = 0; nt < 3; ++nt) {
    v8f acc = {};
    acc = wmma32(a, ldfrag(frags, F_N0W1 + nt, lane), acc);
    const int nn = nt * 16 + (lane & 15);
    const float bb = (nn < HID_F) ? b1[nn] : 0.f;
#pragma unroll
    for (int r = 0; r < 8; ++r) {
      const int mm = r + khalf;
      stage[wave][mm][nn] = (_Float16)lrelu(acc[r] + bb);
    }
  }
  __syncthreads();
  // Layer 2: K = 34 (2 K tiles), N = 20 (2 N tiles).
  v16h y0, y1;
  stage_read(stage[wave], m, khalf, y0, y1);
#pragma unroll
  for (int nt = 0; nt < 2; ++nt) {
    v8f acc = {};
    acc = wmma32(y0, ldfrag(frags, F_N0W2 + nt, lane), acc);
    acc = wmma32(y1, ldfrag(frags, F_N0W2 + 2 + nt, lane), acc);
    const int nn = nt * 16 + (lane & 15);
    const float bb = (nn < REC_F) ? b2[nn] : 0.f;
#pragma unroll
    for (int r = 0; r < 8; ++r) {
      const int mm = r + khalf;
      const int node = base + mm;
      if (wv && nn < REC_F && node < n_nodes)
        h[node * REC_F + nn] = lrelu(acc[r] + bb);
    }
  }
}

// ---------------------------------------------------------------------------
// Edge kernel (bucket d): msg = leaky([bond || h_child] @ inner_w + b),
// atomicAdd into inner_sum[parent]. K = 23 (1 tile), N = 20 (2 tiles).
// ---------------------------------------------------------------------------
__global__ void __launch_bounds__(128) edge_kernel(
    const float* __restrict__ bond, const float* __restrict__ h,
    const int* __restrict__ child, const int* __restrict__ parent,
    const _Float16* __restrict__ frags, const float* __restrict__ ib,
    const int* __restrict__ ebkt, const int* __restrict__ ectr, int d,
    float* __restrict__ inner_sum) {
  const int cnt = ectr[d];
  const int off = ectr[16 + d];
  const int tiles = (cnt + 15) >> 4;
  const int lane = threadIdx.x & 31;
  const int wave = threadIdx.x >> 5;
  const int tile = blockIdx.x * 4 + wave;
  if (tile >= tiles) return;  // wave-uniform; no LDS sync in this kernel
  const int base = tile << 4;
  const int m = lane & 15;
  const int khalf = (lane >> 4) << 3;
  const int ridx = imin(base + m, cnt - 1);
  const int e = ebkt[off + ridx];
  const int ch = child[e];

  v16h a;
#pragma unroll
  for (int j = 0; j < 16; ++j) {
    const int k = khalf + j + ((j & 8) ? 8 : 0);
    float v = 0.f;
    if (k < BOND_F)              v = bond[e * BOND_F + k];
    else if (k < BOND_F + REC_F) v = h[ch * REC_F + (k - BOND_F)];
    a[j] = (_Float16)v;
  }
#pragma unroll
  for (int nt = 0; nt < 2; ++nt) {
    v8f acc = {};
    acc = wmma32(a, ldfrag(frags, F_INNER + nt, lane), acc);
    const int nn = nt * 16 + (lane & 15);
    const float bb = (nn < INNER_F) ? ib[nn] : 0.f;
#pragma unroll
    for (int r = 0; r < 8; ++r) {
      const int mm = r + khalf;
      const int rowi = base + mm;
      if (nn < INNER_F && rowi < cnt) {
        const int ee = ebkt[off + rowi];
        atomicAdd(&inner_sum[parent[ee] * INNER_F + nn], lrelu(acc[r] + bb));
      }
    }
  }
}

// ---------------------------------------------------------------------------
// Node kernel (bucket d): h[node] = leaky(leaky([atom||inner_sum]@w1+b1)@w2+b2)
// K = 34 (2 tiles), layer1 N = 34 (3 tiles), layer2 N = 20 (2 tiles).
// ---------------------------------------------------------------------------
__global__ void __launch_bounds__(128) node_kernel(
    const float* __restrict__ atom, const float* __restrict__ inner_sum,
    const _Float16* __restrict__ frags, const float* __restrict__ b1,
    const float* __restrict__ b2, const int* __restrict__ nbkt,
    const int* __restrict__ nctr, int d, float* __restrict__ h) {
  __shared__ _Float16 stage[4][16][STG_W];
  const int cnt = nctr[d];
  const int off = nctr[16 + d];
  const int tiles = (cnt + 15) >> 4;
  if ((int)blockIdx.x * 4 >= tiles) return;  // uniform block exit
  const int lane = threadIdx.x & 31;
  const int wave = threadIdx.x >> 5;
  const int tile = blockIdx.x * 4 + wave;
  const bool wv = tile < tiles;
  const int base = (wv ? tile : 0) << 4;
  const int m = lane & 15;
  const int khalf = (lane >> 4) << 3;
  const int ridx = imin(base + m, (cnt > 0 ? cnt : 1) - 1);
  const int node = nbkt[off + ridx];

  v16h a0, a1;
#pragma unroll
  for (int j = 0; j < 16; ++j) {
    const int k = khalf + j + ((j & 8) ? 8 : 0);  // 0..31
    const float v0 = (k < ATOM_F) ? atom[node * ATOM_F + k]
                                  : inner_sum[node * INNER_F + (k - ATOM_F)];
    a0[j] = (_Float16)v0;
    const int k1 = 32 + k;
    a1[j] = (_Float16)((k1 < ATOM_F + INNER_F)
                           ? inner_sum[node * INNER_F + (k1 - ATOM_F)] : 0.f);
  }
  stage_pad_zero(stage[wave], lane);
#pragma unroll
  for (int nt = 0; nt < 3; ++nt) {
    v8f acc = {};
    acc = wmma32(a0, ldfrag(frags, F_NW1 + nt, lane), acc);
    acc = wmma32(a1, ldfrag(frags, F_NW1 + 3 + nt, lane), acc);
    const int nn = nt * 16 + (lane & 15);
    const float bb = (nn < HID_F) ? b1[nn] : 0.f;
#pragma unroll
    for (int r = 0; r < 8; ++r) {
      const int mm = r + khalf;
      stage[wave][mm][nn] = (_Float16)lrelu(acc[r] + bb);
    }
  }
  __syncthreads();
  v16h y0, y1;
  stage_read(stage[wave], m, khalf, y0, y1);
#pragma unroll
  for (int nt = 0; nt < 2; ++nt) {
    v8f acc = {};
    acc = wmma32(y0, ldfrag(frags, F_NW2 + nt, lane), acc);
    acc = wmma32(y1, ldfrag(frags, F_NW2 + 2 + nt, lane), acc);
    const int nn = nt * 16 + (lane & 15);
    const float bb = (nn < REC_F) ? b2[nn] : 0.f;
#pragma unroll
    for (int r = 0; r < 8; ++r) {
      const int mm = r + khalf;
      const int rowi = base + mm;
      if (wv && nn < REC_F && rowi < cnt) {
        const int nd = nbkt[off + rowi];
        h[nd * REC_F + nn] = lrelu(acc[r] + bb);
      }
    }
  }
}

// ---------------------------------------------------------------------------
// Root segment-sum and graph readout
// ---------------------------------------------------------------------------
__global__ void root_scatter(const float* __restrict__ h, const float* __restrict__ is_root,
                             const int* __restrict__ gid, float* __restrict__ g, int N) {
  const int n = blockIdx.x * blockDim.x + threadIdx.x;
  if (n >= N) return;
  const float r = is_root[n];
  if (r == 0.f) return;
  const int gg = gid[n];
#pragma unroll
  for (int j = 0; j < REC_F; ++j)
    atomicAdd(&g[gg * REC_F + j], h[n * REC_F + j] * r);
}

__global__ void out_kernel(const float* __restrict__ g, const float* __restrict__ w1,
                           const float* __restrict__ b1, const float* __restrict__ w2,
                           const float* __restrict__ b2, float* __restrict__ out, int G) {
  const int i = blockIdx.x * blockDim.x + threadIdx.x;
  if (i >= G) return;
  float gi[REC_F];
#pragma unroll
  for (int k = 0; k < REC_F; ++k) gi[k] = g[i * REC_F + k];
  float acc = b2[0];
#pragma unroll 5
  for (int j = 0; j < OHID_F; ++j) {
    float t = b1[j];
#pragma unroll
    for (int k = 0; k < REC_F; ++k) t += gi[k] * w1[k * OHID_F + j];
    acc += tanhf(t) * w2[j];
  }
  out[i] = acc;
}

// ---------------------------------------------------------------------------
// Host orchestration
// ---------------------------------------------------------------------------
static inline int cdiv(int a, int b) { return (a + b - 1) / b; }

extern "C" void kernel_launch(void* const* d_in, const int* in_sizes, int n_in,
                              void* d_out, int out_size, void* d_ws, size_t ws_size,
                              hipStream_t stream) {
  const float* atom    = (const float*)d_in[0];
  const float* bond    = (const float*)d_in[1];
  const int*   parent  = (const int*)d_in[2];
  const int*   child   = (const int*)d_in[3];
  const int*   depth   = (const int*)d_in[4];
  const int*   gid     = (const int*)d_in[5];
  const float* is_root = (const float*)d_in[6];
  const float* inner_w = (const float*)d_in[7];
  const float* inner_b = (const float*)d_in[8];
  const float* net_w1  = (const float*)d_in[9];
  const float* net_b1  = (const float*)d_in[10];
  const float* net_w2  = (const float*)d_in[11];
  const float* net_b2  = (const float*)d_in[12];
  const float* net0_w1 = (const float*)d_in[13];
  const float* net0_b1 = (const float*)d_in[14];
  const float* net0_w2 = (const float*)d_in[15];
  const float* net0_b2 = (const float*)d_in[16];
  const float* out_w1  = (const float*)d_in[17];
  const float* out_b1  = (const float*)d_in[18];
  const float* out_w2  = (const float*)d_in[19];
  const float* out_b2  = (const float*)d_in[20];
  float* out = (float*)d_out;

  const int N = in_sizes[4];  // node_depth -> N_NODES
  const int E = in_sizes[2];  // parent_idx -> N_EDGES
  const int G = out_size;     // N_GRAPHS

  // Workspace carve-up (256B aligned slices).
  char* wsb = (char*)d_ws;
  size_t off = 0;
  auto take = [&](size_t bytes) -> void* {
    void* p = wsb + off;
    off += (bytes + 255) & ~(size_t)255;
    return p;
  };
  float*    h     = (float*)take((size_t)N * REC_F * sizeof(float));
  float*    isum  = (float*)take((size_t)N * INNER_F * sizeof(float));
  float*    g     = (float*)take((size_t)G * REC_F * sizeof(float));
  int*      hc    = (int*)take((size_t)N * sizeof(int));
  int*      ebkt  = (int*)take((size_t)E * sizeof(int));
  int*      nbkt  = (int*)take((size_t)N * sizeof(int));
  int*      ectr  = (int*)take(48 * sizeof(int));
  int*      nctr  = (int*)take(48 * sizeof(int));
  _Float16* frags = (_Float16*)take((size_t)N_FRAGS * 32 * 16 * sizeof(_Float16));
  (void)ws_size; (void)n_in;

  // 1) Zero state used this call (isum accumulates; counters/cursors restart).
  zero_f32<<<2048, 256, 0, stream>>>(isum, (long long)N * INNER_F);
  zero_f32<<<256,  256, 0, stream>>>(g, (long long)G * REC_F);
  zero_i32<<<2048, 256, 0, stream>>>(hc, (long long)N);
  zero_i32<<<1, 96, 0, stream>>>(ectr, 48);
  zero_i32<<<1, 96, 0, stream>>>(nctr, 48);

  // 2) Bake weight tiles into wave32 WMMA B-fragment images.
  pack_frags<<<N_FRAGS, 32, 0, stream>>>(inner_w, net0_w1, net0_w2, net_w1,
                                         net_w2, frags);

  // 3) Depth bucketing (each edge/node processed exactly once overall).
  mark_has_child<<<cdiv(E, 256), 256, 0, stream>>>(parent, hc, E);
  count_edges<<<cdiv(E, 256), 256, 0, stream>>>(parent, depth, ectr, E);
  count_nodes<<<cdiv(N, 256), 256, 0, stream>>>(depth, hc, nctr, N);
  prefix16<<<1, 1, 0, stream>>>(ectr);
  prefix16<<<1, 1, 0, stream>>>(nctr);
  scatter_edges<<<cdiv(E, 256), 256, 0, stream>>>(parent, depth, ectr, ebkt, E);
  scatter_nodes<<<cdiv(N, 256), 256, 0, stream>>>(depth, hc, nctr, nbkt, N);

  // 4) Leaf vectors for every node (WMMA).
  leaf_kernel<<<cdiv(N, 64), 128, 0, stream>>>(atom, frags, net0_b1, net0_b2, h, N);

  // 5) Bottom-up sweep, deepest parents first (WMMA edge + node kernels).
  const int egrid = cdiv(E, 64);
  const int ngrid = cdiv(N, 64);
  for (int d = MAXD - 2; d >= 0; --d) {
    edge_kernel<<<egrid, 128, 0, stream>>>(bond, h, child, parent, frags,
                                           inner_b, ebkt, ectr, d, isum);
    node_kernel<<<ngrid, 128, 0, stream>>>(atom, isum, frags, net_b1, net_b2,
                                           nbkt, nctr, d, h);
  }

  // 6) Per-graph root reduction and readout.
  root_scatter<<<cdiv(N, 256), 256, 0, stream>>>(h, is_root, gid, g, N);
  out_kernel<<<cdiv(G, 256), 256, 0, stream>>>(g, out_w1, out_b1, out_w2, out_b2,
                                               out, G);
}
